// CuboidSelfAttentionLayer_71451075937127
// MI455X (gfx1250) — matrix operand order repaired and checked by reading, not compiled, not run on hardware
//
#include <hip/hip_runtime.h>

typedef __attribute__((ext_vector_type(16))) _Float16 v16h;
typedef __attribute__((ext_vector_type(8)))  _Float16 v8h;
typedef __attribute__((ext_vector_type(8)))  float    v8f;
typedef __attribute__((ext_vector_type(4)))  float    v4f;

#define NC_PER_B   512          // (16/2)*(56/7)*(56/7)
#define VOL        98
#define VOLP       112          // padded to 7 WMMA M-tiles
#define C_DIM      128
#define NHEADS     4
#define HD         32

// ---------------- workspace layout (bytes) ----------------
#define WS_A_OFF      0
#define WS_A_BYTES    (1024ull * VOL * C_DIM * 2)           // 25,690,112
#define WS_WQKV_OFF   (WS_A_OFF + WS_A_BYTES)
#define WS_WQKV_BYTES (384ull * 128 * 2)                    // 98,304
#define WS_WPROJ_OFF  (WS_WQKV_OFF + WS_WQKV_BYTES)

// ---------------- LDS layout (bytes), dynamic ----------------
#define L_A_OFF    0                         // [112][128] f16, reused as O
#define L_Q_OFF    28672                     // [112][128] f16
#define L_K_OFF    57344                     // [112][128] f16
#define L_VT_OFF   86016                     // [128][128] f16 (V transposed per head)
#define L_SP_OFF   118784                    // [112][128] f16 (scores then probs, in place)
#define L_TOTAL    147456

// =============== gfx1250 async copy helpers ===============
// GLOBAL_LOAD_ASYNC_TO_LDS_B128, GVS mode: mem = SGPR64 + VGPR32; VDST = LDS byte addr
__device__ __forceinline__ void async_g2l_b128(int lds_addr, unsigned voff,
                                               unsigned long long gbase) {
    asm volatile("global_load_async_to_lds_b128 %0, %1, %2 offset:0"
                 :: "v"(lds_addr), "v"(voff), "s"(gbase) : "memory");
}
__device__ __forceinline__ void wait_async0() {
#if __has_builtin(__builtin_amdgcn_s_wait_asynccnt)
    __builtin_amdgcn_s_wait_asynccnt(0);
#else
    asm volatile("s_wait_asynccnt 0" ::: "memory");
#endif
}

// =============== WMMA helpers (layouts per ISA 7.12.2) ===============
__device__ __forceinline__ v8f wmma16x16x32(v16h a, v16h b, v8f c) {
    return __builtin_amdgcn_wmma_f32_16x16x32_f16(
        /*neg_a=*/false, a, /*neg_b=*/false, b,
        /*c_mod=*/(short)0, c, /*reuse_a=*/false, /*reuse_b=*/false);
}

__device__ __forceinline__ v16h cat16(v8h lo, v8h hi) {
    v16h r;
#pragma unroll
    for (int i = 0; i < 8; ++i) { r[i] = lo[i]; r[i + 8] = hi[i]; }
    return r;
}

// A-matrix (MxK, 16x32 f16): lane L holds row m0+(L&15);
// halves 0..7 = K kb..kb+7, halves 8..15 = K kb+16..kb+23, kb = k0 + (L>=16 ? 8 : 0)
template <typename PTR>
__device__ __forceinline__ v16h load_frag_a(PTR base, int stride, int m0, int k0, int lane) {
    int row = m0 + (lane & 15);
    int kb  = k0 + ((lane >> 4) << 3);
    const _Float16* p = base + row * stride + kb;
    v8h lo = *(const v8h*)(p);
    v8h hi = *(const v8h*)(p + 16);
    return cat16(lo, hi);
}

// B-matrix (KxN, 32x16 f16) supplied from an N-major (transposed, NxK) buffer:
// lane L holds column n0+(L&15), halves 0..7 = K kb..kb+7, halves 8..15 = K kb+16..kb+23, kb = k0 + (L>=16 ? 8 : 0)
template <typename PTR>
__device__ __forceinline__ v16h load_frag_b(PTR base, int stride, int n0, int k0, int lane) {
    int col = n0 + (lane & 15);
    int kb  = k0 + ((lane >> 4) << 3);
    const _Float16* p = base + col * stride + kb;
    v8h lo = *(const v8h*)(p);
    v8h hi = *(const v8h*)(p + 16);
    return cat16(lo, hi);
}

// shift-mask region class for element v of cuboid (ct,ch,cw); valid for v<98
__device__ __forceinline__ int region_class(int ct, int ch, int cw, int v) {
    int it = v / 49; int r = v % 49; int ih = r / 7; int iw = r % 7;
    int rt = (ct < 7) ? 0 : (it == 0 ? 1 : 2);   // t=14 -> 1, t=15 -> 2
    int rh = (ch < 7) ? 0 : (ih < 4 ? 1 : 2);    // h=49..52 -> 1, 53..55 -> 2
    int rw = (cw < 7) ? 0 : (iw < 4 ? 1 : 2);
    return rt * 9 + rh * 3 + rw;
}

// =============== kernel 0: weights -> f16, transposed to [N][K] ===============
__global__ __launch_bounds__(256) void prep_weights_kernel(
    const float* __restrict__ w_qkv, const float* __restrict__ w_proj,
    _Float16* __restrict__ wqkvT, _Float16* __restrict__ wprojT) {
    int i = blockIdx.x * 256 + threadIdx.x;
    if (i < 384 * 128) {
        int n = i >> 7, k = i & 127;
        wqkvT[i] = (_Float16)w_qkv[k * 384 + n];
    }
    int j = i - 384 * 128;
    if (j >= 0 && j < 128 * 128) {
        int n = j >> 7, k = j & 127;
        wprojT[j] = (_Float16)w_proj[k * 128 + n];
    }
}

// =============== kernel 1: LayerNorm + roll(-1,-3,-3) + cuboid reorder -> f16 A ===============
__global__ __launch_bounds__(256) void ln_reorder_kernel(
    const float* __restrict__ x, const float* __restrict__ gamma,
    const float* __restrict__ beta, _Float16* __restrict__ A) {
    int row  = blockIdx.x * 8 + (threadIdx.x >> 5);   // one wave per 128-ch row
    int lane = threadIdx.x & 31;
    // row -> (b,t,h,w)
    int w  = row % 56; int r1 = row / 56;
    int h  = r1 % 56;  int r2 = r1 / 56;
    int t  = r2 % 16;  int b  = r2 / 16;

    const float* px = x + (size_t)row * C_DIM;
    v4f v = *(const v4f*)(px + lane * 4);

    float s = v[0] + v[1] + v[2] + v[3];
    float s2 = v[0]*v[0] + v[1]*v[1] + v[2]*v[2] + v[3]*v[3];
#pragma unroll
    for (int off = 16; off > 0; off >>= 1) {
        s  += __shfl_xor(s,  off, 32);
        s2 += __shfl_xor(s2, off, 32);
    }
    float mu   = s * (1.0f / 128.0f);
    float var  = s2 * (1.0f / 128.0f) - mu * mu;
    float rstd = __frsqrt_rn(var + 1e-5f);

    // shifted coords (roll by (-1,-3,-3)): element (t,h,w) lands at (t-1, h-3, w-3) mod dims
    int ts = t - 1; if (ts < 0) ts += 16;
    int hs = h - 3; if (hs < 0) hs += 56;
    int ws = w - 3; if (ws < 0) ws += 56;
    int ct = ts >> 1, it = ts & 1;
    int ch = hs / 7,  ih = hs % 7;
    int cw = ws / 7,  iw = ws % 7;
    int n  = (ct * 8 + ch) * 8 + cw;
    int vv = (it * 7 + ih) * 7 + iw;

    _Float16* pa = A + (((size_t)(b * NC_PER_B + n) * VOL + vv) * C_DIM) + lane * 4;
    v4f g = *(const v4f*)(gamma + lane * 4);
    v4f be = *(const v4f*)(beta + lane * 4);
#pragma unroll
    for (int q = 0; q < 4; ++q)
        pa[q] = (_Float16)((v[q] - mu) * rstd * g[q] + be[q]);
}

// =============== kernel 2: fused per-cuboid attention ===============
extern __shared__ char smem[];

__global__ __launch_bounds__(256) void cuboid_attn_kernel(
    const _Float16* __restrict__ A, const _Float16* __restrict__ wqkvT,
    const _Float16* __restrict__ wprojT, const float* __restrict__ b_proj,
    float* __restrict__ out) {
    const int wg   = blockIdx.x;            // 0..1023
    const int b    = wg >> 9;
    const int n    = wg & 511;
    const int ct   = n >> 6, ch = (n >> 3) & 7, cw = n & 7;
    const int tid  = threadIdx.x;
    const int wave = tid >> 5, lane = tid & 31;

    _Float16* sA  = (_Float16*)(smem + L_A_OFF);    // GEMM1 A, later O
    _Float16* sQ  = (_Float16*)(smem + L_Q_OFF);
    _Float16* sK  = (_Float16*)(smem + L_K_OFF);
    _Float16* sVT = (_Float16*)(smem + L_VT_OFF);   // [d(=h*32+dd)][j]
    _Float16* sSP = (_Float16*)(smem + L_SP_OFF);   // scores / probs [112][128]

    // ---- prefetch weight panels toward L2/WGP$ (global_prefetch_b8) ----
    __builtin_prefetch((const char*)wqkvT + tid * 384, 0, 3);
    __builtin_prefetch((const char*)wprojT + tid * 128, 0, 3);

    // ---- stage A rows 0..97 via async global->LDS (ASYNCcnt path, no VGPR round-trip) ----
    {
        unsigned long long gbase = (unsigned long long)(A + (size_t)wg * VOL * C_DIM);
        const int nchunks = VOL * C_DIM * 2 / 16;       // 1568 x 16B
        for (int i = tid; i < nchunks; i += 256)
            async_g2l_b128(L_A_OFF + i * 16, (unsigned)(i * 16), gbase);
        // zero pad rows 98..111 and all of sVT with plain stores meanwhile
        for (int i = tid; i < (VOLP - VOL) * C_DIM / 8; i += 256) {
            v8h z = {};
            *(v8h*)(sA + VOL * C_DIM + i * 8) = z;
        }
        for (int i = tid; i < 128 * 128 / 8; i += 256) {
            v8h z = {};
            *(v8h*)(sVT + i * 8) = z;
        }
        wait_async0();
    }
    __syncthreads();

    // ---- GEMM1: QKV = A(112x128) @ WqkvT'(128x384), 7x24 tiles ----
    for (int t = wave; t < 7 * 24; t += 8) {
        int m0 = (t / 24) * 16, n0 = (t % 24) * 16;
        v8f acc = {};
#pragma unroll
        for (int k0 = 0; k0 < 128; k0 += 32) {
            v16h a  = load_frag_a(sA, C_DIM, m0, k0, lane);
            v16h bf = load_frag_b(wqkvT, C_DIM, n0, k0, lane);
            acc = wmma16x16x32(a, bf, acc);
        }
        int col = n0 + (lane & 15);
        int rb  = m0 + ((lane >> 4) << 3);
        if (col < 128) {
#pragma unroll
            for (int r = 0; r < 8; ++r) sQ[(rb + r) * C_DIM + col] = (_Float16)acc[r];
        } else if (col < 256) {
#pragma unroll
            for (int r = 0; r < 8; ++r) sK[(rb + r) * C_DIM + (col - 128)] = (_Float16)acc[r];
        } else {
#pragma unroll
            for (int r = 0; r < 8; ++r) sVT[(col - 256) * C_DIM + (rb + r)] = (_Float16)acc[r];
        }
    }
    __syncthreads();

    const float scale = 0.17677669529663687f;  // 32^-0.5

    for (int h = 0; h < NHEADS; ++h) {
        // ---- GEMM2: S = Q_h(112x32) @ K_h^T(32x112), 7x7 tiles, one WMMA each ----
        for (int t = wave; t < 49; t += 8) {
            int m0 = (t / 7) * 16, j0 = (t % 7) * 16;
            v16h a  = load_frag_a(sQ + h * HD, C_DIM, m0, 0, lane);
            v16h bf = load_frag_b(sK + h * HD, C_DIM, j0, 0, lane);  // B[d][j] = K[j][d]
            v8f acc = {};
            acc = wmma16x16x32(a, bf, acc);
            int col = j0 + (lane & 15);
            int rb  = m0 + ((lane >> 4) << 3);
#pragma unroll
            for (int r = 0; r < 8; ++r) sSP[(rb + r) * C_DIM + col] = (_Float16)acc[r];
        }
        __syncthreads();

        // ---- masked softmax, one wave per row, in place f16 ----
        for (int i = wave; i < VOLP; i += 8) {
            int ci = region_class(ct, ch, cw, i < VOL ? i : 0);
            float vals[4]; bool ok[4];
            float mx = -1e30f;
#pragma unroll
            for (int q = 0; q < 4; ++q) {
                int j = lane * 4 + q;           // 0..127
                float sv = (float)sSP[i * C_DIM + j];
                bool good = (i < VOL) && (j < VOL) &&
                            (region_class(ct, ch, cw, j < VOL ? j : 0) == ci);
                ok[q] = good;
                vals[q] = good ? sv * scale : -1e30f;
                mx = fmaxf(mx, vals[q]);
            }
#pragma unroll
            for (int off = 16; off > 0; off >>= 1) mx = fmaxf(mx, __shfl_xor(mx, off, 32));
            float e[4]; float sum = 0.f;
#pragma unroll
            for (int q = 0; q < 4; ++q) {
                e[q] = ok[q] ? __expf(vals[q] - mx) : 0.f;
                sum += e[q];
            }
#pragma unroll
            for (int off = 16; off > 0; off >>= 1) sum += __shfl_xor(sum, off, 32);
            float inv = (sum > 0.f) ? 1.f / sum : 0.f;
#pragma unroll
            for (int q = 0; q < 4; ++q)
                sSP[i * C_DIM + lane * 4 + q] = (_Float16)(e[q] * inv);
        }
        __syncthreads();

        // ---- GEMM3: O_h = P(112x128) @ V_h(128x32), 7x2 tiles ----
        for (int t = wave; t < 14; t += 8) {
            int m0 = (t >> 1) * 16, n0 = (t & 1) * 16;
            v8f acc = {};
#pragma unroll
            for (int k0 = 0; k0 < 128; k0 += 32) {
                v16h a  = load_frag_a(sSP, C_DIM, m0, k0, lane);
                v16h bf = load_frag_b(sVT + h * HD * C_DIM, C_DIM, n0, k0, lane); // B[j][d]=VT[d][j]
                acc = wmma16x16x32(a, bf, acc);
            }
            int col = n0 + (lane & 15);
            int rb  = m0 + ((lane >> 4) << 3);
#pragma unroll
            for (int r = 0; r < 8; ++r)
                sA[(rb + r) * C_DIM + h * HD + col] = (_Float16)acc[r];  // sA reused as O
        }
        __syncthreads();
    }

    // ---- GEMM4: Y = O(112x128) @ WprojT'(128x128) + bias, scatter to output ----
    for (int t = wave; t < 7 * 8; t += 8) {
        int m0 = (t / 8) * 16, n0 = (t % 8) * 16;
        v8f acc = {};
#pragma unroll
        for (int k0 = 0; k0 < 128; k0 += 32) {
            v16h a  = load_frag_a(sA, C_DIM, m0, k0, lane);
            v16h bf = load_frag_b(wprojT, C_DIM, n0, k0, lane);
            acc = wmma16x16x32(a, bf, acc);
        }
        int col  = n0 + (lane & 15);
        int rb   = m0 + ((lane >> 4) << 3);
        float bias = b_proj[col];
#pragma unroll
        for (int r = 0; r < 8; ++r) {
            int i = rb + r;
            if (i < VOL) {
                int it = i / 49; int rr = i % 49; int ih = rr / 7; int iw = rr % 7;
                int ts = ct * 2 + it, hs = ch * 7 + ih, ws = cw * 7 + iw;
                int to = ts + 1; if (to >= 16) to -= 16;     // roll back (+1,+3,+3)
                int ho = hs + 3; if (ho >= 56) ho -= 56;
                int wo = ws + 3; if (wo >= 56) wo -= 56;
                size_t o = (((size_t)b * 16 + to) * 56 + ho) * 56 + wo;
                out[o * C_DIM + col] = acc[r] + bias;
            }
        }
    }
}

// =============== launcher ===============
extern "C" void kernel_launch(void* const* d_in, const int* in_sizes, int n_in,
                              void* d_out, int out_size, void* d_ws, size_t ws_size,
                              hipStream_t stream) {
    const float* x      = (const float*)d_in[0];
    const float* gamma  = (const float*)d_in[1];
    const float* beta   = (const float*)d_in[2];
    const float* w_qkv  = (const float*)d_in[3];
    const float* w_proj = (const float*)d_in[4];
    const float* b_proj = (const float*)d_in[5];
    float* out = (float*)d_out;

    char* ws = (char*)d_ws;
    _Float16* A      = (_Float16*)(ws + WS_A_OFF);
    _Float16* wqkvT  = (_Float16*)(ws + WS_WQKV_OFF);
    _Float16* wprojT = (_Float16*)(ws + WS_WPROJ_OFF);

    prep_weights_kernel<<<(384 * 128 + 128 * 128 + 255) / 256, 256, 0, stream>>>(
        w_qkv, w_proj, wqkvT, wprojT);

    ln_reorder_kernel<<<(2 * 16 * 56 * 56) / 8, 256, 0, stream>>>(x, gamma, beta, A);

    cuboid_attn_kernel<<<1024, 256, L_TOTAL, stream>>>(A, wqkvT, wprojT, b_proj, out);
}
